// ResonantAudioModel_61435212202477
// MI455X (gfx1250) — matrix operand order repaired and checked by reading, not compiled, not run on hardware
//
#include <hip/hip_runtime.h>
#include <cstdint>

// Problem constants (match reference)
#define Bsz   16
#define Tlen  4096
#define INF_  80
#define Hd    256
#define Ld    8
#define D2d   128
#define ROWS  (Bsz * Tlen)        // 65536
#define CHN   (Bsz * D2d)         // 2048 scan channels
#define NCHUNK 32
#define CLEN  (Tlen / NCHUNK)     // 128

typedef __attribute__((ext_vector_type(16))) _Float16    v16h;
typedef __attribute__((ext_vector_type(8)))  float       v8f;
typedef __attribute__((ext_vector_type(4)))  unsigned int u32x4;

union FragAB { v16h h; u32x4 q[2]; };

// LDS row pitch in halves: 32 data + 8 pad => 80B stride => rows land on
// distinct LDS banks for the 16-lane b128 fragment loads (20*k mod 64 distinct).
#define KP 40

// ---------------------------------------------------------------------------
// Generic WMMA GEMM:  C[M x N] = A[M x K] @ B[K x N] + bias (+ C if BETA)
// Block = 256 threads = 8 waves; each wave owns a 16-row M tile and all NT
// N-tiles. K is staged through LDS in 32-wide slices as f16 (transposed B).
// ---------------------------------------------------------------------------
template<int NT, int BETA>
__global__ __launch_bounds__(256)
void gemm_wmma_kernel(const float* __restrict__ A, int lda,
                      const float* __restrict__ Bm,     // K x N row-major
                      const float* __restrict__ bias,   // [N]
                      float* __restrict__ C,
                      int N, int K)
{
    __shared__ _Float16 sA[128 * KP];
    __shared__ _Float16 sB[256 * KP];

    const int tid  = threadIdx.x;
    const int wave = tid >> 5;
    const int lane = tid & 31;
    const int rowBase = blockIdx.x * 128;

    const int lrow = (wave << 4) + (lane & 15);        // A-frag source row in sA
    const int koff = (lane >> 4) << 3;                 // A-frag: 0 or 8
    const int kb   = (lane >> 4) << 3;                 // B-frag: 0 or 8
    const int mrow = rowBase + (wave << 4) + ((lane >> 4) << 3); // C row for r=0

    v8f acc[NT];
    #pragma unroll
    for (int nt = 0; nt < NT; ++nt) {
        const int col = (nt << 4) + (lane & 15);
        #pragma unroll
        for (int r = 0; r < 8; ++r)
            acc[nt][r] = BETA ? C[(size_t)(mrow + r) * N + col] : 0.0f;
    }

    const int nK = (K + 31) >> 5;
    for (int kc = 0; kc < nK; ++kc) {
        const int k0 = kc << 5;

        // Stage A tile (128 x 32), f32 -> f16, zero-pad past K
        for (int e = tid; e < 128 * 32; e += 256) {
            int rr = e >> 5, cc = e & 31;
            int gk = k0 + cc;
            float v = (gk < K) ? A[(size_t)(rowBase + rr) * lda + gk] : 0.0f;
            sA[rr * KP + cc] = (_Float16)v;
        }
        // Stage B tile (32 x N) transposed -> sB[n][k]
        for (int e = tid; e < 32 * NT * 16; e += 256) {
            int kk = e / (NT * 16), nn = e % (NT * 16);
            int gk = k0 + kk;
            float v = (gk < K) ? Bm[(size_t)gk * N + nn] : 0.0f;
            sB[nn * KP + kk] = (_Float16)v;
        }
        // Hint next A slice into cache (global_prefetch_b8)
        if (kc + 1 < nK && tid < 128)
            __builtin_prefetch(&A[(size_t)(rowBase + tid) * lda + k0 + 32], 0, 0);

        __syncthreads();

        // A fragment: lanes 0-15 -> K koff..koff+7 and koff+16..koff+23
        FragAB fa;
        fa.q[0] = *(const u32x4*)(sA + lrow * KP + koff);
        fa.q[1] = *(const u32x4*)(sA + lrow * KP + koff + 16);

        #pragma unroll
        for (int nt = 0; nt < NT; ++nt) {
            const int nb = ((nt << 4) + (lane & 15)) * KP;
            FragAB fb;   // lanes 0-15: K 0..7,16..23 ; lanes 16-31: K 8..15,24..31
            fb.q[0] = *(const u32x4*)(sB + nb + kb);
            fb.q[1] = *(const u32x4*)(sB + nb + kb + 16);
            acc[nt] = __builtin_amdgcn_wmma_f32_16x16x32_f16(
                false, fa.h, false, fb.h, (short)0, acc[nt], false, false);
        }
        __syncthreads();
    }

    #pragma unroll
    for (int nt = 0; nt < NT; ++nt) {
        const int col = (nt << 4) + (lane & 15);
        const float bv = bias[col];
        #pragma unroll
        for (int r = 0; r < 8; ++r)
            C[(size_t)(mrow + r) * N + col] = acc[nt][r] + bv;
    }
}

// ---------------------------------------------------------------------------
// Chunked parallel scan of  h_t = c * h_{t-1} + u_t  (c complex, diagonal).
// UH holds u on entry, h on exit (in place). 3 phases.
// ---------------------------------------------------------------------------
__device__ __forceinline__ void channel_coeff(const float* dl, const float* th,
                                              int d, float& ca, float& cb)
{
    float r = 1.0f / (1.0f + __expf(-dl[d]));
    ca = r * __cosf(th[d]);
    cb = r * __sinf(th[d]);
}

__global__ __launch_bounds__(256)
void scan_phaseA(float* __restrict__ uh,
                 const float* __restrict__ dl, const float* __restrict__ th,
                 float* __restrict__ carry)
{
    int gid   = blockIdx.x * 256 + threadIdx.x;   // CHN * NCHUNK threads
    int ch    = gid & (CHN - 1);
    int chunk = gid >> 11;
    int b = ch >> 7, d = ch & (D2d - 1);
    float ca, cb; channel_coeff(dl, th, d, ca, cb);

    float hre = 0.0f, him = 0.0f;
    size_t base = ((size_t)b * Tlen + (size_t)chunk * CLEN) * Hd + d;
    for (int j = 0; j < CLEN; ++j) {
        float ure = uh[base];
        float uim = uh[base + D2d];
        float nr = __builtin_fmaf(ca, hre, __builtin_fmaf(-cb, him, ure));
        float ni = __builtin_fmaf(cb, hre, __builtin_fmaf( ca, him, uim));
        uh[base] = nr; uh[base + D2d] = ni;
        hre = nr; him = ni;
        base += Hd;
    }
    size_t ci = 2 * ((size_t)chunk * CHN + ch);
    carry[ci] = hre; carry[ci + 1] = him;
}

__global__ __launch_bounds__(256)
void scan_phaseB(float* __restrict__ carry,
                 const float* __restrict__ dl, const float* __restrict__ th)
{
    int ch = blockIdx.x * 256 + threadIdx.x;      // CHN threads
    int d = ch & (D2d - 1);
    float pr, pi; channel_coeff(dl, th, d, pr, pi);
    #pragma unroll
    for (int s = 0; s < 7; ++s) {                 // c^128 via repeated squaring
        float nr = pr * pr - pi * pi;
        float ni = 2.0f * pr * pi;
        pr = nr; pi = ni;
    }
    float ar = 0.0f, ai = 0.0f;
    for (int p = 0; p < NCHUNK; ++p) {
        size_t ci = 2 * ((size_t)p * CHN + ch);
        float hr = carry[ci], hi = carry[ci + 1];
        float nr = hr + pr * ar - pi * ai;
        float ni = hi + pi * ar + pr * ai;
        carry[ci] = nr; carry[ci + 1] = ni;       // now true h at chunk end
        ar = nr; ai = ni;
    }
}

__global__ __launch_bounds__(256)
void scan_phaseC(float* __restrict__ uh,
                 const float* __restrict__ dl, const float* __restrict__ th,
                 const float* __restrict__ carry)
{
    int gid   = blockIdx.x * 256 + threadIdx.x;   // CHN * (NCHUNK-1) threads
    int ch    = gid & (CHN - 1);
    int chunk = (gid >> 11) + 1;                  // chunk 0 needs no fix-up
    int b = ch >> 7, d = ch & (D2d - 1);
    float ca, cb; channel_coeff(dl, th, d, ca, cb);

    size_t ci = 2 * ((size_t)(chunk - 1) * CHN + ch);
    float h0r = carry[ci], h0i = carry[ci + 1];
    float fr = ca, fi = cb;                       // c^1
    size_t base = ((size_t)b * Tlen + (size_t)chunk * CLEN) * Hd + d;
    for (int j = 0; j < CLEN; ++j) {
        uh[base]       += fr * h0r - fi * h0i;
        uh[base + D2d] += fi * h0r + fr * h0i;
        float nfr = fr * ca - fi * cb;
        float nfi = fr * cb + fi * ca;
        fr = nfr; fi = nfi;
        base += Hd;
    }
}

__global__ __launch_bounds__(256)
void copy_state_kernel(const float* __restrict__ uh, float* __restrict__ st)
{
    int gid = blockIdx.x * 256 + threadIdx.x;     // B*H = 4096 threads
    int b = gid >> 8;
    int hcol = gid & (Hd - 1);
    st[gid] = uh[((size_t)b * Tlen + (Tlen - 1)) * Hd + hcol];
}

// ---------------------------------------------------------------------------
extern "C" void kernel_launch(void* const* d_in, const int* in_sizes, int n_in,
                              void* d_out, int out_size, void* d_ws, size_t ws_size,
                              hipStream_t stream)
{
    (void)in_sizes; (void)n_in; (void)out_size; (void)ws_size;

    const float* x    = (const float*)d_in[0];
    const float* Win  = (const float*)d_in[1];
    const float* bin  = (const float*)d_in[2];
    const float* Wout = (const float*)d_in[3];
    const float* bout = (const float*)d_in[4];
    const float* Wu   = (const float*)d_in[5];
    const float* bu   = (const float*)d_in[6];
    const float* Wy   = (const float*)d_in[7];
    const float* by   = (const float*)d_in[8];
    const float* dl   = (const float*)d_in[9];
    const float* th   = (const float*)d_in[10];

    float* X     = (float*)d_ws;                   // ROWS*Hd f32 (xt, in place)
    float* UH    = X  + (size_t)ROWS * Hd;         // ROWS*Hd f32 (u -> h, in place)
    float* carry = UH + (size_t)ROWS * Hd;         // CHN*NCHUNK*2 f32

    float* y     = (float*)d_out;                  // B*T*IN
    float* state = y + (size_t)ROWS * INF_;        // L*B*H

    dim3 blk(256);
    dim3 gemmGrid(ROWS / 128);

    // X = x @ W_in + b_in
    gemm_wmma_kernel<16, 0><<<gemmGrid, blk, 0, stream>>>(x, INF_, Win, bin, X, Hd, INF_);

    for (int l = 0; l < Ld; ++l) {
        const float* dli = dl + l * D2d;
        const float* thi = th + l * D2d;
        // U = X @ Wu[l] + bu[l]
        gemm_wmma_kernel<16, 0><<<gemmGrid, blk, 0, stream>>>(
            X, Hd, Wu + (size_t)l * Hd * Hd, bu + (size_t)l * Hd, UH, Hd, Hd);
        // diagonal complex scan over T (in place, U -> H)
        scan_phaseA<<<dim3(CHN * NCHUNK / 256), blk, 0, stream>>>(UH, dli, thi, carry);
        scan_phaseB<<<dim3(CHN / 256),          blk, 0, stream>>>(carry, dli, thi);
        scan_phaseC<<<dim3(CHN * (NCHUNK - 1) / 256), blk, 0, stream>>>(UH, dli, thi, carry);
        // final state for this layer (h at t = T-1)
        copy_state_kernel<<<dim3(Bsz * Hd / 256), blk, 0, stream>>>(
            UH, state + (size_t)l * Bsz * Hd);
        // X += H @ Wy[l] + by[l]
        gemm_wmma_kernel<16, 1><<<gemmGrid, blk, 0, stream>>>(
            UH, Hd, Wy + (size_t)l * Hd * Hd, by + (size_t)l * Hd, X, Hd, Hd);
    }

    // y = X @ W_out + b_out
    gemm_wmma_kernel<5, 0><<<gemmGrid, blk, 0, stream>>>(X, Hd, Wout, bout, y, INF_, Hd);
}
